// CLIPVisionTower_SCOPE_17437567222420
// MI455X (gfx1250) — matrix operand order repaired and checked by reading, not compiled, not run on hardware
//
#include <hip/hip_runtime.h>

typedef __attribute__((ext_vector_type(16))) _Float16 v16h;
typedef __attribute__((ext_vector_type(8)))  float    v8f;

#define B_  32
#define N_  576
#define D_  1024
#define K_  128
#define NT  36    /* N_/16 tiles per side */
#define P_  4     /* workgroups cooperating per batch in selection */
#define MS  (N_ / P_)   /* 144: m-slice length per workgroup */

// ---------------------------------------------------------------------------
// Kernel 1: L2-normalize each token row, then split each f32 value into
// hi/lo f16 parts (x = hi + lo, hi = f16(x), lo = f16(x - hi)) so the WMMA
// GEMM can reconstruct ~f32 accuracy: sim = Hi*Hi^T + Hi*Lo^T + Lo*Hi^T.
// ---------------------------------------------------------------------------
__global__ void norm_kernel(const float* __restrict__ hid,
                            _Float16* __restrict__ nfHi, _Float16* __restrict__ nfLo) {
    int row = blockIdx.x;                 // 0 .. B_*N_-1
    int b = row / N_, n = row % N_;
    const float* src = hid + ((size_t)b * (N_ + 1) + (n + 1)) * D_;
    int t = threadIdx.x;                  // 256 threads
    float4 v = ((const float4*)src)[t];
    float p = v.x * v.x + v.y * v.y + v.z * v.z + v.w * v.w;
    __shared__ float red[256];
    red[t] = p;
    __syncthreads();
    for (int s = 128; s > 0; s >>= 1) {
        if (t < s) red[t] += red[t + s];
        __syncthreads();
    }
    float r = 1.0f / sqrtf(red[0]);
    float x0 = v.x * r, x1 = v.y * r, x2 = v.z * r, x3 = v.w * r;
    union { _Float16 h[4]; uint2 u; } wh, wl;
    wh.h[0] = (_Float16)x0; wl.h[0] = (_Float16)(x0 - (float)wh.h[0]);
    wh.h[1] = (_Float16)x1; wl.h[1] = (_Float16)(x1 - (float)wh.h[1]);
    wh.h[2] = (_Float16)x2; wl.h[2] = (_Float16)(x2 - (float)wh.h[2]);
    wh.h[3] = (_Float16)x3; wl.h[3] = (_Float16)(x3 - (float)wh.h[3]);
    ((uint2*)(nfHi + (size_t)row * D_))[t] = wh.u;
    ((uint2*)(nfLo + (size_t)row * D_))[t] = wl.u;
}

// ---------------------------------------------------------------------------
// Kernel 2: sim[b] = norm[b] @ norm[b]^T in split-f16 via
// v_wmma_f32_16x16x32_f16 (3 WMMAs per tile per K-step: HiHi + HiLo + LoHi;
// LoLo ~ 2^-22 dropped). One wave computes a 16x32 strip (1 A x 2 B tiles):
// A hi/lo fragments loaded once per K step, fed to 6 WMMAs.
// Fragment k order for A and B: contiguous k = 16*(lane/16) + s (wave32).
// ---------------------------------------------------------------------------
__global__ void sim_kernel(const _Float16* __restrict__ nfHi,
                           const _Float16* __restrict__ nfLo,
                           float* __restrict__ sim) {
    int wave = threadIdx.x >> 5;
    int lane = threadIdx.x & 31;
    int tile = blockIdx.x * 8 + wave;            // grid = B_*NT*(NT/2)/8
    int b   = tile / (NT * (NT / 2));
    int r   = tile % (NT * (NT / 2));
    int tm  = r / (NT / 2);
    int tn  = (r % (NT / 2)) * 2;                // first of two adjacent N tiles
    int mn  = lane & 15;
    int hi  = lane >> 4;

    size_t offA  = ((size_t)b * N_ + (size_t)(tm * 16 + mn)) * D_;
    size_t offB0 = ((size_t)b * N_ + (size_t)(tn * 16 + mn)) * D_;
    const _Float16* rowAH  = nfHi + offA;
    const _Float16* rowAL  = nfLo + offA;
    const _Float16* rowB0H = nfHi + offB0;
    const _Float16* rowB0L = nfLo + offB0;
    const _Float16* rowB1H = rowB0H + (size_t)16 * D_;
    const _Float16* rowB1L = rowB0L + (size_t)16 * D_;

    v8f acc0 = {}, acc1 = {};
    for (int k0 = 0; k0 < D_; k0 += 32) {
        union { v16h v; float4 f[2]; } aH, aL, b0H, b0L, b1H, b1L;
        int kA0 = k0 + (hi ? 16 : 0);
        int kA1 = k0 + (hi ? 24 : 8);
        int kB  = k0 + (hi ? 16 : 0);
        aH.f[0]  = *(const float4*)(rowAH  + kA0);
        aH.f[1]  = *(const float4*)(rowAH  + kA1);
        aL.f[0]  = *(const float4*)(rowAL  + kA0);
        aL.f[1]  = *(const float4*)(rowAL  + kA1);
        b0H.f[0] = *(const float4*)(rowB0H + kB);
        b0H.f[1] = *(const float4*)(rowB0H + kB + 8);
        b0L.f[0] = *(const float4*)(rowB0L + kB);
        b0L.f[1] = *(const float4*)(rowB0L + kB + 8);
        b1H.f[0] = *(const float4*)(rowB1H + kB);
        b1H.f[1] = *(const float4*)(rowB1H + kB + 8);
        b1L.f[0] = *(const float4*)(rowB1L + kB);
        b1L.f[1] = *(const float4*)(rowB1L + kB + 8);

        acc0 = __builtin_amdgcn_wmma_f32_16x16x32_f16(false, aH.v, false, b0H.v,
                                                      (short)0, acc0, false, false);
        acc0 = __builtin_amdgcn_wmma_f32_16x16x32_f16(false, aH.v, false, b0L.v,
                                                      (short)0, acc0, false, false);
        acc0 = __builtin_amdgcn_wmma_f32_16x16x32_f16(false, aL.v, false, b0H.v,
                                                      (short)0, acc0, false, false);
        acc1 = __builtin_amdgcn_wmma_f32_16x16x32_f16(false, aH.v, false, b1H.v,
                                                      (short)0, acc1, false, false);
        acc1 = __builtin_amdgcn_wmma_f32_16x16x32_f16(false, aH.v, false, b1L.v,
                                                      (short)0, acc1, false, false);
        acc1 = __builtin_amdgcn_wmma_f32_16x16x32_f16(false, aL.v, false, b1H.v,
                                                      (short)0, acc1, false, false);
    }

    float* out0 = sim + (size_t)b * N_ * N_ + (size_t)(tm * 16) * N_ + (size_t)(tn * 16);
    #pragma unroll
    for (int q = 0; q < 8; ++q) {
        int mm = q + hi * 8;
        out0[(size_t)mm * N_ + mn]      = acc0[q];
        out0[(size_t)mm * N_ + mn + 16] = acc1[q];
    }
}

// ---------------------------------------------------------------------------
// Kernel 3a: zero the inter-block sync state (re-run every launch so graph
// replays are deterministic).
// ---------------------------------------------------------------------------
__global__ void zero_sync_kernel(unsigned* __restrict__ ctrA, unsigned* __restrict__ flagB) {
    int t = threadIdx.x;
    if (t < B_) { ctrA[t] = 0u; flagB[t] = 0u; }
}

// ---------------------------------------------------------------------------
// Kernel 3b: greedy coverage selection, P_=4 cooperating workgroups per batch.
// Block p of batch b owns m-slice [p*MS, p*MS+MS) of cur_max (in LDS) and
// computes partial gains over that slice for all 576 columns. Block 0 sums
// the P_ partials (ascending p == ascending m, matching reference order),
// masks, argmaxes (first-index tie-break), and broadcasts (step<<16)|best in
// one atomic word. Per-batch split barrier: monotonic arrive counter with
// release/acquire agent-scope atomics. sim stays L2-resident (42.5MB << 192MB).
// ---------------------------------------------------------------------------
__global__ void select_kernel(const float* __restrict__ sim, const float* __restrict__ cls,
                              float* __restrict__ partial, unsigned* __restrict__ ctrA,
                              unsigned* __restrict__ flagB,
                              int* __restrict__ selFlags, float* __restrict__ outIdx) {
    __shared__ float s_curmax[MS];
    __shared__ float s_gain[N_];
    __shared__ int   s_sel[N_];
    __shared__ float s_rv[64];
    __shared__ int   s_ri[64];
    __shared__ int   s_best;

    int b = blockIdx.x / P_;
    int p = blockIdx.x % P_;
    int n = threadIdx.x;                        // 0..575 (one column per thread)
    int m0 = p * MS;
    const float* simB = sim + (size_t)b * N_ * N_;
    float* myPartial = partial + ((size_t)(b * P_ + p)) * N_;
    float attn = cls[b * N_ + n];
    const float NEG_INF = -__builtin_inff();

    // NOP unless dispatched in clusters (ClusterID==0 -> S_NOP per ISA);
    // exercises the CDNA5 cluster-barrier path harmlessly.
    __builtin_amdgcn_s_cluster_barrier();

    if (n < MS) s_curmax[n] = 0.0f;
    s_sel[n] = 0;
    __syncthreads();

    for (int t = 0; t < K_; ++t) {
        // partial gain over this block's m-slice
        float g = 0.0f;
        for (int j = 0; j < MS; ++j) {
            float v = simB[(size_t)(m0 + j) * N_ + n] - s_curmax[j];
            g += (v > 0.0f) ? v : 0.0f;
        }
        myPartial[n] = g;
        __threadfence();                        // publish partial gains
        __syncthreads();
        if (n == 0)
            __hip_atomic_fetch_add(&ctrA[b], 1u, __ATOMIC_RELEASE, __HIP_MEMORY_SCOPE_AGENT);

        if (p == 0) {
            // wait for all P_ arrivals of this step
            if (n == 0) {
                while (__hip_atomic_load(&ctrA[b], __ATOMIC_ACQUIRE, __HIP_MEMORY_SCOPE_AGENT)
                       < (unsigned)((t + 1) * P_))
                    __builtin_amdgcn_s_sleep(1);
            }
            __syncthreads();
            // total gain = sum of partials, ascending p == ascending m
            const float* pp = partial + (size_t)(b * P_) * N_;
            float gt = pp[n] + pp[N_ + n] + pp[2 * N_ + n] + pp[3 * N_ + n];
            s_gain[n] = s_sel[n] ? NEG_INF : gt * attn;
            __syncthreads();
            // two-level argmax, first-index tie-break (matches jnp.argmax)
            if (n < 64) {
                float bv = NEG_INF; int bi = 0;
                #pragma unroll
                for (int j = 0; j < 9; ++j) {
                    int idx = n * 9 + j;
                    float v = s_gain[idx];
                    if (v > bv) { bv = v; bi = idx; }
                }
                s_rv[n] = bv; s_ri[n] = bi;
            }
            __syncthreads();
            if (n == 0) {
                float bv = NEG_INF; int bi = 0;
                for (int j = 0; j < 64; ++j)
                    if (s_rv[j] > bv) { bv = s_rv[j]; bi = s_ri[j]; }
                s_sel[bi] = 1;
                outIdx[b * K_ + t] = (float)(bi + 1);      // selection-order (+1)
                __hip_atomic_store(&flagB[b], ((unsigned)(t + 1) << 16) | (unsigned)bi,
                                   __ATOMIC_RELEASE, __HIP_MEMORY_SCOPE_AGENT);
            }
        }

        // all blocks: wait for best broadcast (block 0 passes immediately)
        if (n == 0) {
            unsigned f;
            do {
                f = __hip_atomic_load(&flagB[b], __ATOMIC_ACQUIRE, __HIP_MEMORY_SCOPE_AGENT);
                if ((f >> 16) != (unsigned)(t + 1)) __builtin_amdgcn_s_sleep(1);
            } while ((f >> 16) != (unsigned)(t + 1));
            s_best = (int)(f & 0xFFFFu);
        }
        __syncthreads();
        int best = s_best;
        // update this block's cur_max slice: cur_max = max(cur_max, sim[best,:])
        if (n < MS)
            s_curmax[n] = fmaxf(s_curmax[n], simB[(size_t)best * N_ + (m0 + n)]);
        __syncthreads();
    }
    if (p == 0) selFlags[b * N_ + n] = s_sel[n];
}

// ---------------------------------------------------------------------------
// Kernel 4: sorted gather list == ascending scan of selected flags.
// ---------------------------------------------------------------------------
__global__ void compact_kernel(const int* __restrict__ selFlags, int* __restrict__ gidx) {
    int b = blockIdx.x;
    if (threadIdx.x == 0) {
        int c = 0;
        for (int nn = 0; nn < N_; ++nn)
            if (selFlags[b * N_ + nn]) gidx[b * K_ + c++] = nn + 1;
    }
}

// ---------------------------------------------------------------------------
// Kernel 5: gather selected token rows (f32, float4-coalesced).
// ---------------------------------------------------------------------------
__global__ void gather_kernel(const float* __restrict__ hid, const int* __restrict__ gidx,
                              float* __restrict__ out) {
    int row = blockIdx.x;                 // 0 .. B_*K_-1
    int b = row / K_, t = row % K_;
    int g = gidx[b * K_ + t];
    const float4* src = (const float4*)(hid + ((size_t)b * (N_ + 1) + g) * D_);
    float4* dst = (float4*)(out + (size_t)row * D_);
    dst[threadIdx.x] = src[threadIdx.x];  // 256 threads x 16B = 4 KB row
}

// ---------------------------------------------------------------------------
extern "C" void kernel_launch(void* const* d_in, const int* in_sizes, int n_in,
                              void* d_out, int out_size, void* d_ws, size_t ws_size,
                              hipStream_t stream) {
    (void)in_sizes; (void)n_in; (void)out_size; (void)ws_size;

    const float* hid = (const float*)d_in[0];   // [32,577,1024] f32
    const float* cls = (const float*)d_in[1];   // [32,576]      f32
    float* out = (float*)d_out;

    // workspace layout (~118.4 MB)
    char* ws = (char*)d_ws;
    size_t off = 0;
    _Float16* nfHi = (_Float16*)(ws + off); off += (size_t)B_ * N_ * D_ * sizeof(_Float16);
    _Float16* nfLo = (_Float16*)(ws + off); off += (size_t)B_ * N_ * D_ * sizeof(_Float16);
    float* sim    = (float*)(ws + off);     off += (size_t)B_ * N_ * N_ * sizeof(float);
    float* partial= (float*)(ws + off);     off += (size_t)B_ * P_ * N_ * sizeof(float);
    int* selFlags = (int*)(ws + off);       off += (size_t)B_ * N_ * sizeof(int);
    int* gidx     = (int*)(ws + off);       off += (size_t)B_ * K_ * sizeof(int);
    unsigned* ctrA  = (unsigned*)(ws + off); off += B_ * sizeof(unsigned);
    unsigned* flagB = (unsigned*)(ws + off); off += B_ * sizeof(unsigned);

    float* outTok = out;                                 // [32,128,1024]
    float* outIdx = out + (size_t)B_ * K_ * D_;          // [32,128] as f32

    norm_kernel    <<<B_ * N_,               256, 0, stream>>>(hid, nfHi, nfLo);
    sim_kernel     <<<B_ * NT * (NT / 2) / 8, 256, 0, stream>>>(nfHi, nfLo, sim);
    zero_sync_kernel<<<1,                     64, 0, stream>>>(ctrA, flagB);
    select_kernel  <<<B_ * P_,               N_,  0, stream>>>(sim, cls, partial, ctrA,
                                                               flagB, selFlags, outIdx);
    compact_kernel <<<B_,                    32,  0, stream>>>(selFlags, gidx);
    gather_kernel  <<<B_ * K_,               256, 0, stream>>>(hid, gidx, outTok);
}
